// RecurrentRGCN_50276887167213
// MI455X (gfx1250) — matrix operand order repaired and checked by reading, not compiled, not run on hardware
//
#include <hip/hip_runtime.h>
#include <hip/hip_bf16.h>

#define N_NODES 20000
#define N_EDGES 400000
#define H 200
#define HP 224                 // K padded to 7*32
#define KT 7                   // k-tiles of 32
#define NTILES_M 1250          // 20000 / 16
#define NTILES_N 13            // ceil(200/16)
#define WPK_ELEMS (KT * NTILES_N * 32 * 16)   // packed weight elements
#define RRELU_SLOPE 0.22916666666666666f      // 11/48

typedef __attribute__((ext_vector_type(16))) __bf16 v16bf;
typedef __attribute__((ext_vector_type(8)))  __bf16 v8bf;
typedef __attribute__((ext_vector_type(8)))  float  v8f;

// ---------------- in-degree ----------------
__global__ void deg_kernel(const int* __restrict__ dst, float* __restrict__ deg) {
    int e = blockIdx.x * blockDim.x + threadIdx.x;
    if (e < N_EDGES) atomicAdd(&deg[dst[e]], 1.0f);
}

__global__ void normmask_kernel(const float* __restrict__ deg,
                                float* __restrict__ norm, float* __restrict__ mask) {
    int n = blockIdx.x * blockDim.x + threadIdx.x;
    if (n < N_NODES) {
        float d = deg[n];
        norm[n] = 1.0f / fmaxf(d, 1.0f);
        mask[n] = (d > 0.0f) ? 1.0f : 0.0f;
    }
}

// ---------------- h0 = ent_emb[node_id] ----------------
__global__ void gather_kernel(const float* __restrict__ ent,
                              const int* __restrict__ node_id,
                              float* __restrict__ h0) {
    long idx = (long)blockIdx.x * blockDim.x + threadIdx.x;
    if (idx < (long)N_NODES * H) {
        int n = (int)(idx / H);
        int j = (int)(idx % H);
        h0[idx] = ent[(size_t)node_id[n] * H + j];
    }
}

// ---------------- pre[dst] += h[src] + rel[etype] (segment-sum BEFORE matmul) ----
__global__ void scatter_kernel(const float* __restrict__ h,
                               const float* __restrict__ rel,
                               const int* __restrict__ src,
                               const int* __restrict__ dst,
                               const int* __restrict__ et,
                               float* __restrict__ pre) {
    long idx = (long)blockIdx.x * blockDim.x + threadIdx.x;
    const long total = (long)N_EDGES * (H / 4);
    if (idx >= total) return;
    int e = (int)(idx / (H / 4));
    int c = (int)(idx % (H / 4)) * 4;
    int s = src[e], d = dst[e], t = et[e];
    const float4 hv = *(const float4*)(h   + (size_t)s * H + c);
    const float4 rv = *(const float4*)(rel + (size_t)t * H + c);
    float* p = pre + (size_t)d * H + c;
    atomicAdd(p + 0, hv.x + rv.x);
    atomicAdd(p + 1, hv.y + rv.y);
    atomicAdd(p + 2, hv.z + rv.z);
    atomicAdd(p + 3, hv.w + rv.w);
}

// ---------------- pack node matrices to padded bf16 (norm folded into preb) ----
__global__ void pack_nodes_kernel(const float* __restrict__ pre,
                                  const float* __restrict__ h,
                                  const float* __restrict__ norm,
                                  __bf16* __restrict__ preb,
                                  __bf16* __restrict__ hb) {
    long idx = (long)blockIdx.x * blockDim.x + threadIdx.x;
    if (idx >= (long)N_NODES * HP) return;
    int m = (int)(idx / HP);
    int k = (int)(idx % HP);
    float pv = 0.0f, hv = 0.0f;
    if (k < H) {
        pv = pre[(size_t)m * H + k] * norm[m];
        hv = h[(size_t)m * H + k];
    }
    preb[idx] = (__bf16)pv;
    hb[idx]   = (__bf16)hv;
}

// ---------------- pack 200x200 f32 weights into B-fragment bf16 layout ----------
// Wp[((kt*NTILES_N + tn)*32 + lane)*16 + e] = W[kt*32 + 8*(lane>>4) + (e&7) + 16*(e>>3)][tn*16 + (lane&15)]
__global__ void pack_w_kernel(const float* __restrict__ W, __bf16* __restrict__ Wp) {
    int idx = blockIdx.x * blockDim.x + threadIdx.x;
    if (idx >= WPK_ELEMS) return;
    int e    = idx & 15;
    int lane = (idx >> 4) & 31;
    int t    = idx >> 9;            // kt*NTILES_N + tn
    int tn   = t % NTILES_N;
    int kt   = t / NTILES_N;
    int k = kt * 32 + 8 * (lane >> 4) + (e & 7) + 16 * (e >> 3);
    int n = tn * 16 + (lane & 15);
    Wp[idx] = (k < H && n < H) ? (__bf16)W[(size_t)k * H + n] : (__bf16)0.0f;
}

// ---------------- fused dense layer ----------------
// out = rrelu( (pre*norm)@Wn + (mask ? h@Wl : h@We) )
// 8 waves/block, one 16x16 tile per wave, fully unrolled K, no guarded loads.
__global__ __launch_bounds__(256)
void rgcn_mm_kernel(const __bf16* __restrict__ preb,   // [N][HP] bf16, norm folded
                    const __bf16* __restrict__ hb,     // [N][HP] bf16
                    const __bf16* __restrict__ Wnp,    // packed B-frag layout
                    const __bf16* __restrict__ Wlp,
                    const __bf16* __restrict__ Wep,
                    const float*  __restrict__ mask,
                    float* __restrict__ out) {
    const int lane = threadIdx.x;                 // 0..31
    const int tn   = blockIdx.x;                  // 0..12
    const int tm   = blockIdx.y * 8 + threadIdx.y;
    if (tm >= NTILES_M) return;                   // whole wave exits together
    const int hi = lane >> 4;
    const int ln = lane & 15;
    const int m  = tm * 16 + ln;                  // A row for this lane
    const int n  = tn * 16 + ln;                  // B col for this lane

    const __bf16* ap_row = preb + (size_t)m * HP;
    const __bf16* ah_row = hb   + (size_t)m * HP;
    const __bf16* wn_base = Wnp + (size_t)tn * (32 * 16);
    const __bf16* wl_base = Wlp + (size_t)tn * (32 * 16);
    const __bf16* we_base = Wep + (size_t)tn * (32 * 16);

    v8f accA = {}, accL = {}, accE = {};
#pragma unroll
    for (int kt = 0; kt < KT; ++kt) {
        const int kk = kt * 32;
        // A-frag: lane holds K = kk + 8*hi + [0..7] and kk + 16 + 8*hi + [0..7]
        v8bf p_lo = *(const v8bf*)(ap_row + kk + 8 * hi);
        v8bf p_hi = *(const v8bf*)(ap_row + kk + 16 + 8 * hi);
        v8bf h_lo = *(const v8bf*)(ah_row + kk + 8 * hi);
        v8bf h_hi = *(const v8bf*)(ah_row + kk + 16 + 8 * hi);
        v16bf ap = __builtin_shufflevector(p_lo, p_hi, 0,1,2,3,4,5,6,7,8,9,10,11,12,13,14,15);
        v16bf ah = __builtin_shufflevector(h_lo, h_hi, 0,1,2,3,4,5,6,7,8,9,10,11,12,13,14,15);
        // B-frags: one contiguous 32B per lane from packed layout
        const size_t woff = ((size_t)kt * NTILES_N * 32 + lane) * 16;
        v16bf bn = *(const v16bf*)(wn_base + woff);
        v16bf bl = *(const v16bf*)(wl_base + woff);
        v16bf be = *(const v16bf*)(we_base + woff);

        accA = __builtin_amdgcn_wmma_f32_16x16x32_bf16(false, ap, false, bn, (short)0, accA, false, false);
        accL = __builtin_amdgcn_wmma_f32_16x16x32_bf16(false, ah, false, bl, (short)0, accL, false, false);
        accE = __builtin_amdgcn_wmma_f32_16x16x32_bf16(false, ah, false, be, (short)0, accE, false, false);
    }

    if (n < H) {
#pragma unroll
        for (int i = 0; i < 8; ++i) {             // C/D layout: row = i + 8*hi
            int row = tm * 16 + 8 * hi + i;
            float loopv = (mask[row] > 0.0f) ? accL[i] : accE[i];
            float v = accA[i] + loopv;
            v = (v >= 0.0f) ? v : v * RRELU_SLOPE;
            out[(size_t)row * H + n] = v;
        }
    }
}

extern "C" void kernel_launch(void* const* d_in, const int* in_sizes, int n_in,
                              void* d_out, int out_size, void* d_ws, size_t ws_size,
                              hipStream_t stream) {
    const float* ent  = (const float*)d_in[0];
    const float* rel  = (const float*)d_in[1];
    const float* W1[3] = { (const float*)d_in[2], (const float*)d_in[3], (const float*)d_in[4] };
    const float* W2[3] = { (const float*)d_in[5], (const float*)d_in[6], (const float*)d_in[7] };
    const int* node_id = (const int*)d_in[8];
    const int* src     = (const int*)d_in[9];
    const int* dst     = (const int*)d_in[10];
    const int* etype   = (const int*)d_in[11];
    float* out = (float*)d_out;

    const size_t NH  = (size_t)N_NODES * H;
    const size_t NHP = (size_t)N_NODES * HP;

    char* base = (char*)d_ws;
    float* h0   = (float*)base;  base += NH * sizeof(float);
    float* h1   = (float*)base;  base += NH * sizeof(float);
    float* pre  = (float*)base;  base += NH * sizeof(float);
    float* deg  = (float*)base;  base += N_NODES * sizeof(float);
    float* norm = (float*)base;  base += N_NODES * sizeof(float);
    float* mask = (float*)base;  base += N_NODES * sizeof(float);
    __bf16* preb = (__bf16*)base; base += NHP * sizeof(__bf16);
    __bf16* hb   = (__bf16*)base; base += NHP * sizeof(__bf16);
    __bf16* Wp[6];
    for (int i = 0; i < 6; ++i) { Wp[i] = (__bf16*)base; base += WPK_ELEMS * sizeof(__bf16); }

    // degrees / norm / mask (shared by both layers)
    hipMemsetAsync(deg, 0, N_NODES * sizeof(float), stream);
    deg_kernel<<<(N_EDGES + 255) / 256, 256, 0, stream>>>(dst, deg);
    normmask_kernel<<<(N_NODES + 255) / 256, 256, 0, stream>>>(deg, norm, mask);

    // pack the six weight matrices once
    const int wblocks = (WPK_ELEMS + 255) / 256;
    for (int i = 0; i < 3; ++i) {
        pack_w_kernel<<<wblocks, 256, 0, stream>>>(W1[i], Wp[i]);
        pack_w_kernel<<<wblocks, 256, 0, stream>>>(W2[i], Wp[3 + i]);
    }

    // h0 = ent_emb[node_id]
    gather_kernel<<<(int)((NH + 255) / 256), 256, 0, stream>>>(ent, node_id, h0);

    const long scatter_items  = (long)N_EDGES * (H / 4);
    const int  scatter_blocks = (int)((scatter_items + 255) / 256);
    const int  pack_blocks    = (int)((NHP + 255) / 256);
    dim3 mm_block(32, 8);
    dim3 mm_grid(NTILES_N, (NTILES_M + 7) / 8);

    // ---- layer 1 ----
    hipMemsetAsync(pre, 0, NH * sizeof(float), stream);
    scatter_kernel<<<scatter_blocks, 256, 0, stream>>>(h0, rel, src, dst, etype, pre);
    pack_nodes_kernel<<<pack_blocks, 256, 0, stream>>>(pre, h0, norm, preb, hb);
    rgcn_mm_kernel<<<mm_grid, mm_block, 0, stream>>>(preb, hb, Wp[0], Wp[1], Wp[2], mask, h1);

    // ---- layer 2 ----
    hipMemsetAsync(pre, 0, NH * sizeof(float), stream);
    scatter_kernel<<<scatter_blocks, 256, 0, stream>>>(h1, rel, src, dst, etype, pre);
    pack_nodes_kernel<<<pack_blocks, 256, 0, stream>>>(pre, h1, norm, preb, hb);
    rgcn_mm_kernel<<<mm_grid, mm_block, 0, stream>>>(preb, hb, Wp[3], Wp[4], Wp[5], mask, out);
}
